// CausalMultiheadSelfAttention_62405874811930
// MI455X (gfx1250) — compile-verified, matrix operand order checked by reading, not run on hardware
//
#include <hip/hip_runtime.h>
#include <hip/hip_bf16.h>
#include <math.h>

// Problem constants (from reference)
#define LSEQ 1024
#define NB   4
#define CH   2048
#define NH   16
#define DH   128
#define WIN  512
#define SCALE 0.08838834764831845f  // 1/sqrt(128)

typedef _Float16 half_t;
typedef __attribute__((ext_vector_type(16))) _Float16 v16h;
typedef __attribute__((ext_vector_type(8)))  _Float16 v8h;
typedef __attribute__((ext_vector_type(8)))  float    v8f;

// ---------------------------------------------------------------------------
// WMMA helpers (CDNA5 wave32; layouts per CDNA5 ISA §7.12.2)
//
// 16-bit A-matrix 16x32: lane l holds row m = l&15; h = l>>4 selects which
// K-halves: frag[0..7]  <- A[m][k0 + 8h .. k0+8h+7]
//           frag[8..15] <- A[m][k0 + 16 + 8h .. k0+16+8h+7]
// B-matrix 32x16 mirrors with lane = column n (lane holds one row of the
// *transposed* B, contiguous along K) -> same loader for both operands.
// ---------------------------------------------------------------------------
static __device__ inline v16h ld_frag(const half_t* __restrict__ base, int ld,
                                      int row, int k0, int h) {
    const half_t* p = base + (size_t)row * ld + k0 + h * 8;
    v8h lo = *(const v8h*)(p);
    v8h hi = *(const v8h*)(p + 16);
    v16h r;
#pragma unroll
    for (int i = 0; i < 8; ++i) { r[i] = lo[i]; r[i + 8] = hi[i]; }
    return r;
}

static __device__ inline v8f wmma16(v16h a, v16h b, v8f c) {
    return __builtin_amdgcn_wmma_f32_16x16x32_f16(
        false, a, false, b, (short)0, c, false, false);
}

// ---------------------------------------------------------------------------
// Prep: x (L,N,C) f32 -> xb (N,L,C) f16 ; generic f32->f16 weight convert
// ---------------------------------------------------------------------------
__global__ void cvt_x_kernel(const float* __restrict__ x, half_t* __restrict__ xb) {
    size_t gid = (size_t)blockIdx.x * blockDim.x + threadIdx.x;
    if (gid >= (size_t)LSEQ * NB * CH) return;
    int c = gid % CH;
    int n = (gid / CH) % NB;
    int l = gid / ((size_t)CH * NB);
    xb[((size_t)n * LSEQ + l) * CH + c] = (half_t)x[gid];
}

__global__ void cvt_w_kernel(const float* __restrict__ w, half_t* __restrict__ wh, int count) {
    int gid = blockIdx.x * blockDim.x + threadIdx.x;
    if (gid < count) wh[gid] = (half_t)w[gid];
}

// ---------------------------------------------------------------------------
// Software-pipelined 16x64 GEMM body: acc[c] += A(16xK) * B(c)(Kx16)
// Double-buffered fragments: loads for step k+32 are issued before the WMMAs
// for step k, so v_wmma waits become partial instead of s_wait_loadcnt 0.
// ---------------------------------------------------------------------------
static __device__ inline void gemm_strip_16x64(
    const half_t* __restrict__ A, const half_t* __restrict__ B,
    int nl, int hi, v8f acc[4]) {
    v16h a0, a1, b0[4], b1[4];
    a0 = ld_frag(A, CH, nl, 0, hi);
#pragma unroll
    for (int c = 0; c < 4; ++c) b0[c] = ld_frag(B + (size_t)(c * 16) * CH, CH, nl, 0, hi);

    for (int k = 0; k < CH; k += 64) {
        // stage k+32 while computing k
        a1 = ld_frag(A, CH, nl, k + 32, hi);
#pragma unroll
        for (int c = 0; c < 4; ++c)
            b1[c] = ld_frag(B + (size_t)(c * 16) * CH, CH, nl, k + 32, hi);
        __builtin_prefetch(A + (size_t)nl * CH + k + 128, 0, 1);
#pragma unroll
        for (int c = 0; c < 4; ++c) acc[c] = wmma16(a0, b0[c], acc[c]);

        // stage k+64 (wrap to 0 on the tail so no OOB; result unused then)
        int kn = (k + 64 < CH) ? (k + 64) : 0;
        a0 = ld_frag(A, CH, nl, kn, hi);
#pragma unroll
        for (int c = 0; c < 4; ++c)
            b0[c] = ld_frag(B + (size_t)(c * 16) * CH, CH, nl, kn, hi);
#pragma unroll
        for (int c = 0; c < 4; ++c) acc[c] = wmma16(a1, b1[c], acc[c]);
    }
}

// ---------------------------------------------------------------------------
// QKV GEMM: qkv[n,l,o] = xb[n,l,:] . W[o,:] + b[o],  o in [0,3C)
// One wave computes a 16(row) x 64(col) strip. 3C/64 = 96 strips per row tile.
// ---------------------------------------------------------------------------
__global__ __launch_bounds__(32) void gemm_qkv_kernel(
    const half_t* __restrict__ xb, const half_t* __restrict__ wq,
    const half_t* __restrict__ wkv, const float* __restrict__ bq,
    const float* __restrict__ bkv, float* __restrict__ qkv) {
    int tile = blockIdx.x;
    int oc = (tile % 96) * 64;
    int t2 = tile / 96;
    int lt = (t2 & 63) * 16;
    int n  = t2 >> 6;

    int lane = threadIdx.x;
    int hi = lane >> 4;
    int nl = lane & 15;

    const half_t* A = xb + ((size_t)n * LSEQ + lt) * CH;
    const half_t* B = (oc < CH) ? (wq + (size_t)oc * CH)
                                : (wkv + (size_t)(oc - CH) * CH);
    v8f acc[4] = {v8f{}, v8f{}, v8f{}, v8f{}};
    gemm_strip_16x64(A, B, nl, hi, acc);

#pragma unroll
    for (int c = 0; c < 4; ++c)
#pragma unroll
        for (int r = 0; r < 8; ++r) {
            int row = lt + r + 8 * hi;
            int col = oc + c * 16 + nl;
            float bias = (col < CH) ? bq[col] : bkv[col - CH];
            qkv[((size_t)n * LSEQ + row) * (3 * CH) + col] = acc[c][r] + bias;
        }
}

// ---------------------------------------------------------------------------
// RoPE + split:  qkv (N,L,3C) f32 -> qh,kh (N,H,L,D) f16 (roped, pos = W+l)
//                               -> vt (N,H,D,L) f16 (transposed for PV B-op)
// One thread per (n,l,head,t) pair, t = d/2.
// ---------------------------------------------------------------------------
__global__ void rope_split_kernel(const float* __restrict__ qkv,
                                  half_t* __restrict__ qh,
                                  half_t* __restrict__ kh,
                                  half_t* __restrict__ vt) {
    int gid = blockIdx.x * blockDim.x + threadIdx.x;  // N*L*H*64 = 4194304
    if (gid >= NB * LSEQ * NH * (DH / 2)) return;
    int t  = gid & 63;
    int hd = (gid >> 6) & 15;
    int l  = (gid >> 10) & (LSEQ - 1);
    int n  = gid >> 20;

    const float* row = qkv + ((size_t)n * LSEQ + l) * (3 * CH);
    float inv = __powf(10000.0f, -(2.0f * t) / (float)DH);
    float ang = (float)(WIN + l) * inv;
    float s, c;
    __sincosf(ang, &s, &c);

    int d0 = hd * DH + 2 * t;
    size_t qb = (((size_t)(n * NH + hd) * LSEQ) + l) * DH + 2 * t;

    float q0 = row[d0], q1 = row[d0 + 1];
    qh[qb]     = (half_t)(q0 * c - q1 * s);
    qh[qb + 1] = (half_t)(q0 * s + q1 * c);

    float k0 = row[CH + d0], k1 = row[CH + d0 + 1];
    kh[qb]     = (half_t)(k0 * c - k1 * s);
    kh[qb + 1] = (half_t)(k0 * s + k1 * c);

    float v0 = row[2 * CH + d0], v1 = row[2 * CH + d0 + 1];
    size_t vb = ((size_t)(n * NH + hd) * DH + 2 * t) * LSEQ + l;
    vt[vb]        = (half_t)v0;
    vt[vb + LSEQ] = (half_t)v1;
}

// ---------------------------------------------------------------------------
// Sliding-window flash attention. One wave per (n, head, 16-query tile).
// Window per row i: j in [i-512, i].  Keys processed in chunks of 32.
// V fragments are issued right after the S WMMAs so the softmax VALU work
// (shuffle reductions + v_exp_f32) hides their latency.
// ---------------------------------------------------------------------------
__global__ __launch_bounds__(32) void attn_kernel(
    const half_t* __restrict__ qh, const half_t* __restrict__ kh,
    const half_t* __restrict__ vt, const float* __restrict__ alibi,
    half_t* __restrict__ attn) {
    int tile = blockIdx.x;               // N*H*(L/16) = 4096
    int it = (tile & 63) * 16;
    int hd = (tile >> 6) & 15;
    int n  = tile >> 10;

    int lane = threadIdx.x;
    int hi = lane >> 4;
    int nl = lane & 15;

    const half_t* Q  = qh + (size_t)(n * NH + hd) * LSEQ * DH;
    const half_t* K  = kh + (size_t)(n * NH + hd) * LSEQ * DH;
    const half_t* Vt = vt + (size_t)(n * NH + hd) * DH * LSEQ;  // (D, L)
    float slope = alibi[hd];

    __shared__ __align__(16) half_t pbuf[16 * 32];

    // Q fragments: 4 chunks over D = 128
    v16h qf[4];
#pragma unroll
    for (int c = 0; c < 4; ++c)
        qf[c] = ld_frag(Q + (size_t)it * DH, DH, nl, c * 32, hi);

    v8f acc[8];
#pragma unroll
    for (int c = 0; c < 8; ++c) acc[c] = v8f{};
    float rmax[8], rsum[8];
#pragma unroll
    for (int r = 0; r < 8; ++r) { rmax[r] = -1e30f; rsum[r] = 0.0f; }

    int jb0 = max(0, it - WIN) & ~31;
    for (int jb = jb0; jb <= it; jb += 32) {
        // ---- S = Q K^T for key columns [jb, jb+32): load all K frags first
        v16h kb0[4], kb1[4];
#pragma unroll
        for (int c = 0; c < 4; ++c) {
            kb0[c] = ld_frag(K + (size_t)jb * DH, DH, nl, c * 32, hi);
            kb1[c] = ld_frag(K + (size_t)(jb + 16) * DH, DH, nl, c * 32, hi);
        }
        v8f s0 = v8f{}, s1 = v8f{};
#pragma unroll
        for (int c = 0; c < 4; ++c) {
            s0 = wmma16(qf[c], kb0[c], s0);
            s1 = wmma16(qf[c], kb1[c], s1);
        }
        // ---- issue V fragments now; softmax below overlaps their latency
        v16h bv[8];
#pragma unroll
        for (int c = 0; c < 8; ++c)
            bv[c] = ld_frag(Vt + (size_t)(c * 16) * LSEQ, LSEQ, nl, jb, hi);

        int j0 = jb + nl;
        int j1 = j0 + 16;
#pragma unroll
        for (int r = 0; r < 8; ++r) {
            int m = it + r + 8 * hi;   // query index
            float e0 = (j0 <= m && j0 >= m - WIN)
                           ? s0[r] * SCALE + slope * (float)(j0 - m) : -1e30f;
            float e1 = (j1 <= m && j1 >= m - WIN)
                           ? s1[r] * SCALE + slope * (float)(j1 - m) : -1e30f;
            // row max over the 16 lanes of this half
            float t = fmaxf(e0, e1);
            t = fmaxf(t, __shfl_xor(t, 1, 32));
            t = fmaxf(t, __shfl_xor(t, 2, 32));
            t = fmaxf(t, __shfl_xor(t, 4, 32));
            t = fmaxf(t, __shfl_xor(t, 8, 32));
            float mn = fmaxf(rmax[r], t);
            float corr = __expf(rmax[r] - mn);
            rmax[r] = mn;
            rsum[r] *= corr;
#pragma unroll
            for (int c = 0; c < 8; ++c) acc[c][r] *= corr;
            float p0 = __expf(e0 - mn);
            float p1 = __expf(e1 - mn);
            float ts = p0 + p1;
            ts += __shfl_xor(ts, 1, 32);
            ts += __shfl_xor(ts, 2, 32);
            ts += __shfl_xor(ts, 4, 32);
            ts += __shfl_xor(ts, 8, 32);
            rsum[r] += ts;
            // D-layout -> LDS (16x32 P tile)
            pbuf[(r + 8 * hi) * 32 + nl]      = (half_t)p0;
            pbuf[(r + 8 * hi) * 32 + 16 + nl] = (half_t)p1;
        }
        __syncthreads();
        // A-layout P fragment from LDS, then PV into 16x128 accumulator
        v16h pa = ld_frag(pbuf, 32, nl, 0, hi);
#pragma unroll
        for (int c = 0; c < 8; ++c) acc[c] = wmma16(pa, bv[c], acc[c]);
        __syncthreads();
    }

    // normalize + store attn (N,L,C) f16, c = head*128 + d
#pragma unroll
    for (int c = 0; c < 8; ++c)
#pragma unroll
        for (int r = 0; r < 8; ++r) {
            int l = it + r + 8 * hi;
            int col = hd * DH + c * 16 + nl;
            attn[((size_t)n * LSEQ + l) * CH + col] =
                (half_t)(acc[c][r] / rsum[r]);
        }
}

// ---------------------------------------------------------------------------
// Output projection: out[l,n,c] = attn[n,l,:] . wo[c,:] + bo[c]
// One wave per 16x64 strip; stores transposed to (L,N,C) f32.
// ---------------------------------------------------------------------------
__global__ __launch_bounds__(32) void gemm_out_kernel(
    const half_t* __restrict__ attn, const half_t* __restrict__ wo,
    const float* __restrict__ bo, float* __restrict__ out) {
    int tile = blockIdx.x;               // N*(L/16)*(C/64) = 8192
    int oc = (tile % 32) * 64;
    int t2 = tile / 32;
    int lt = (t2 & 63) * 16;
    int n  = t2 >> 6;

    int lane = threadIdx.x;
    int hi = lane >> 4;
    int nl = lane & 15;

    const half_t* A = attn + ((size_t)n * LSEQ + lt) * CH;
    const half_t* B = wo + (size_t)oc * CH;
    v8f acc[4] = {v8f{}, v8f{}, v8f{}, v8f{}};
    gemm_strip_16x64(A, B, nl, hi, acc);

#pragma unroll
    for (int c = 0; c < 4; ++c)
#pragma unroll
        for (int r = 0; r < 8; ++r) {
            int l = lt + r + 8 * hi;
            int col = oc + c * 16 + nl;
            out[(size_t)l * NB * CH + (size_t)n * CH + col] = acc[c][r] + bo[col];
        }
}

// ---------------------------------------------------------------------------
extern "C" void kernel_launch(void* const* d_in, const int* in_sizes, int n_in,
                              void* d_out, int out_size, void* d_ws, size_t ws_size,
                              hipStream_t stream) {
    const float* x     = (const float*)d_in[0];
    const float* wq    = (const float*)d_in[1];
    const float* bq    = (const float*)d_in[2];
    const float* wkv   = (const float*)d_in[3];
    const float* bkv   = (const float*)d_in[4];
    const float* wo    = (const float*)d_in[5];
    const float* bo    = (const float*)d_in[6];
    const float* alibi = (const float*)d_in[7];
    float* out = (float*)d_out;

    // workspace carve-up
    float* qkv = (float*)d_ws;                                  // N*L*3C f32
    half_t* xb   = (half_t*)(qkv + (size_t)NB * LSEQ * 3 * CH); // N*L*C
    half_t* wqh  = xb   + (size_t)NB * LSEQ * CH;               // C*C
    half_t* wkvh = wqh  + (size_t)CH * CH;                      // 2C*C
    half_t* woh  = wkvh + (size_t)2 * CH * CH;                  // C*C
    half_t* qh   = woh  + (size_t)CH * CH;                      // N*H*L*D
    half_t* kh   = qh   + (size_t)NB * NH * LSEQ * DH;
    half_t* vt   = kh   + (size_t)NB * NH * LSEQ * DH;
    half_t* attn = vt   + (size_t)NB * NH * LSEQ * DH;          // N*L*C

    // 1) converts
    {
        size_t tot = (size_t)LSEQ * NB * CH;
        cvt_x_kernel<<<(tot + 255) / 256, 256, 0, stream>>>(x, xb);
        cvt_w_kernel<<<(CH * CH + 255) / 256, 256, 0, stream>>>(wq, wqh, CH * CH);
        cvt_w_kernel<<<(2 * CH * CH + 255) / 256, 256, 0, stream>>>(wkv, wkvh, 2 * CH * CH);
        cvt_w_kernel<<<(CH * CH + 255) / 256, 256, 0, stream>>>(wo, woh, CH * CH);
    }
    // 2) QKV projection (WMMA, pipelined)
    gemm_qkv_kernel<<<NB * (LSEQ / 16) * (3 * CH / 64), 32, 0, stream>>>(
        xb, wqh, wkvh, bq, bkv, qkv);
    // 3) RoPE + split into attention layouts
    {
        int tot = NB * LSEQ * NH * (DH / 2);
        rope_split_kernel<<<(tot + 255) / 256, 256, 0, stream>>>(qkv, qh, kh, vt);
    }
    // 4) sliding-window flash attention (WMMA)
    attn_kernel<<<NB * NH * (LSEQ / 16), 32, 0, stream>>>(qh, kh, vt, alibi, attn);
    // 5) output projection + transpose (WMMA, pipelined)
    gemm_out_kernel<<<NB * (LSEQ / 16) * (CH / 64), 32, 0, stream>>>(
        attn, woh, bo, out);
}
